// EncodeProcessDecode_45801531244893
// MI455X (gfx1250) — matrix operand order repaired and checked by reading, not compiled, not run on hardware
//
#include <hip/hip_runtime.h>
#include <cstdint>
#include <cstddef>

// ---------------------------------------------------------------------------
// Types for CDNA5 WMMA (wave32): V_WMMA_F32_16X16X32_BF16
// ---------------------------------------------------------------------------
typedef __attribute__((ext_vector_type(16))) __bf16 v16bf;
typedef __attribute__((ext_vector_type(8)))  float  v8f;

#define GN_NODES 50000
#define GN_EDGES 400000
#define GN_LATENT 64
#define HS_LD 136   // 128 + 8 bf16 pad -> 272B row stride, 16B aligned, spreads LDS banks

struct PieceP { const float* src; const int* idx; int ld; int bcast; };
struct MlpP {
  PieceP piece[4];                 // concat pieces (64-wide each, 32 for encoder)
  const __bf16* wt1; const float* b1;   // wt1: [128][KIN] bf16 (pre-transposed)
  const __bf16* wt2; const float* b2;   // wt2: [NOUT][128] bf16 (zero-padded rows for decoder)
  const float* gamma; const float* beta;
  float* y; int y_ld; int y_cols; int add_inplace;
  float* pool; const int* pool_idx;    // segment_sum target (nullable)
  float* colsum;                       // column-sum target (nullable)
  int M;
};

// ---------------------------------------------------------------------------
// Small utility kernels
// ---------------------------------------------------------------------------
__global__ void zero_f32(float* p, long n) {
  long i = (long)blockIdx.x * blockDim.x + threadIdx.x;
  if (i < n) p[i] = 0.f;
}

// w: f32 [K][Nsrc] row-major  ->  wt: bf16 [Npad][K]  (rows >= Nsrc zero-filled)
__global__ void convert_wt(const float* __restrict__ w, __bf16* __restrict__ wt,
                           int K, int Nsrc, int Npad) {
  int i = blockIdx.x * blockDim.x + threadIdx.x;
  if (i >= Npad * K) return;
  int n = i / K, k = i - n * K;
  float v = (n < Nsrc) ? w[(size_t)k * Nsrc + n] : 0.f;
  wt[i] = (__bf16)v;
}

// ---------------------------------------------------------------------------
// Fragment loaders (layouts per CDNA5 ISA §7.12.2)
// ---------------------------------------------------------------------------
// B frag (32x16 bf16): lane l holds col N=l&15; j<8 -> K = 8*(l>>4) + j ; j>=8 -> K = 8*(l>>4) + 16 + (j-8)
__device__ inline v16bf load_b_frag(const __bf16* wt, int n, int kin, int kb, int hi) {
  const uint4* p = (const uint4*)(wt + (size_t)n * kin + kb + 8 * hi);
  union { uint4 u[2]; v16bf v; } t;
  t.u[0] = p[0]; t.u[1] = p[2];
  return t.v;
}
// A frag (16x32 bf16) from LDS bf16 row: lane l row=l&15;
// j<8 -> K = kb + 8*hi + j ; j>=8 -> K = kb + 8*hi + 16 + (j-8)
__device__ inline v16bf load_a_lds(const __bf16* rowptr, int kb, int hi) {
  const uint4* p0 = (const uint4*)(rowptr + kb + 8 * hi);
  const uint4* p1 = (const uint4*)(rowptr + kb + 8 * hi + 16);
  union { uint4 u[2]; v16bf v; } t;
  t.u[0] = *p0; t.u[1] = *p1;
  return t.v;
}
// A frag gathered from a global f32 row, converting f32->bf16 in registers
__device__ inline v16bf gather_a_f32(const float* rowptr, int kb, int hi) {
  const float* q0 = rowptr + kb + 8 * hi;
  const float* q1 = q0 + 16;
  v16bf a;
#pragma unroll
  for (int j = 0; j < 8; ++j) { a[j] = (__bf16)q0[j]; a[j + 8] = (__bf16)q1[j]; }
  return a;
}

// ---------------------------------------------------------------------------
// Fused MLP: concat-gather -> [16,KIN]x[KIN,128] -> bias+relu -> [16,128]x[128,NOUT]
//            -> bias (+ LayerNorm) (+ colsum atomics) (+ segment-sum atomics)
//            (+ residual in place | store).  One wave owns 16 rows; no barriers.
// ---------------------------------------------------------------------------
template <int KIN, int NOUT, bool LN>
__global__ __launch_bounds__(256) void mlp_kernel(MlpP P) {
  constexpr int NB1 = 8;          // 128 / 16
  constexpr int NB2 = NOUT / 16;  // 4 (latent) or 1 (decoder)
  constexpr int KB1 = KIN / 32;
  __shared__ __bf16 Hs[8][16 * HS_LD];  // per-wave hidden tile, bf16

  const int lane = threadIdx.x & 31;
  const int wave = threadIdx.x >> 5;
  const int nwaves = blockDim.x >> 5;
  const int mbase = (blockIdx.x * nwaves + wave) * 16;
  if (mbase >= P.M) return;
  const int lrow = lane & 15;
  const int hi = lane >> 4;

  // Per-lane row pointers for each concat piece (A layout: lane's row = l&15)
  int row_a = mbase + lrow;
  if (row_a >= P.M) row_a = P.M - 1;  // clamp; tail stores are masked below
  const float* aptr[4];
#pragma unroll
  for (int p = 0; p < 4; ++p) {
    const PieceP pc = P.piece[p];
    if (pc.src) {
      int r = pc.bcast ? 0 : (pc.idx ? pc.idx[row_a] : row_a);
      aptr[p] = pc.src + (size_t)r * pc.ld;
    } else {
      aptr[p] = nullptr;
    }
  }

  v8f zero8;
#pragma unroll
  for (int j = 0; j < 8; ++j) zero8[j] = 0.f;

  // ---- layer 1: X[16,KIN] @ W1[KIN,128] ----
  v8f acc1[NB1];
#pragma unroll
  for (int nb = 0; nb < NB1; ++nb) acc1[nb] = zero8;
#pragma unroll
  for (int kb = 0; kb < KB1; ++kb) {
    const int kg = kb * 32;
    v16bf a = gather_a_f32(aptr[kg >> 6], kg & 63, hi);
#pragma unroll
    for (int nb = 0; nb < NB1; ++nb) {
      v16bf b = load_b_frag(P.wt1, nb * 16 + lrow, KIN, kg, hi);
      acc1[nb] = __builtin_amdgcn_wmma_f32_16x16x32_bf16(
          false, a, false, b, (short)0, acc1[nb], false, false);
    }
  }
  // bias + relu -> bf16 hidden tile in LDS (D layout: row = 8*hi+i, col = 16*nb+lrow)
  __bf16* hs = Hs[wave];
#pragma unroll
  for (int nb = 0; nb < NB1; ++nb) {
    const int col = nb * 16 + lrow;
    const float bias = P.b1[col];
#pragma unroll
    for (int i = 0; i < 8; ++i) {
      float h = acc1[nb][i] + bias;
      h = h > 0.f ? h : 0.f;
      hs[(8 * hi + i) * HS_LD + col] = (__bf16)h;
    }
  }

  // ---- layer 2: H[16,128] @ W2[128,NOUT] ----
  v8f acc2[NB2];
#pragma unroll
  for (int nb = 0; nb < NB2; ++nb) acc2[nb] = zero8;
#pragma unroll
  for (int kb = 0; kb < 4; ++kb) {
    v16bf a = load_a_lds(hs + lrow * HS_LD, kb * 32, hi);
#pragma unroll
    for (int nb = 0; nb < NB2; ++nb) {
      v16bf b = load_b_frag(P.wt2, nb * 16 + lrow, 128, kb * 32, hi);
      acc2[nb] = __builtin_amdgcn_wmma_f32_16x16x32_bf16(
          false, a, false, b, (short)0, acc2[nb], false, false);
    }
  }

  // ---- epilogue (stay in D-fragment layout) ----
  float vals[NB2][8];
#pragma unroll
  for (int nb = 0; nb < NB2; ++nb) {
    const int col = nb * 16 + lrow;
    const float b2 = (col < P.y_cols) ? P.b2[col] : 0.f;
#pragma unroll
    for (int i = 0; i < 8; ++i) vals[nb][i] = acc2[nb][i] + b2;
  }

  if constexpr (LN) {
    // Row 8*hi+i lives across the 16 lanes sharing `hi` (4 cols each):
    // butterfly over lane masks 1,2,4,8 reduces within that group.
#pragma unroll
    for (int i = 0; i < 8; ++i) {
      float s = 0.f, ss = 0.f;
#pragma unroll
      for (int nb = 0; nb < NB2; ++nb) { float x = vals[nb][i]; s += x; ss += x * x; }
#pragma unroll
      for (int m = 1; m < 16; m <<= 1) {
        s += __shfl_xor(s, m, 32);
        ss += __shfl_xor(ss, m, 32);
      }
      const float mu = s * (1.f / NOUT);
      const float var = ss * (1.f / NOUT) - mu * mu;
      const float inv = rsqrtf(var + 1e-3f);
#pragma unroll
      for (int nb = 0; nb < NB2; ++nb) {
        const int col = nb * 16 + lrow;
        vals[nb][i] = (vals[nb][i] - mu) * inv * P.gamma[col] + P.beta[col];
      }
    }
  }

  // column sums of the *new* values (for the context update)
  if (P.colsum) {
#pragma unroll
    for (int nb = 0; nb < NB2; ++nb) {
      float part = 0.f;
#pragma unroll
      for (int i = 0; i < 8; ++i) {
        int r = mbase + 8 * hi + i;
        part += (r < P.M) ? vals[nb][i] : 0.f;
      }
      atomicAdd(&P.colsum[nb * 16 + lrow], part);
    }
  }

  // segment-sum pooling + residual/store
#pragma unroll
  for (int i = 0; i < 8; ++i) {
    const int r = mbase + 8 * hi + i;
    if (r < P.M) {
      if (P.pool) {
        const size_t pr = (size_t)P.pool_idx[r];
#pragma unroll
        for (int nb = 0; nb < NB2; ++nb)
          atomicAdd(&P.pool[pr * GN_LATENT + nb * 16 + lrow], vals[nb][i]);
      }
      float* yrow = P.y + (size_t)r * P.y_ld;
#pragma unroll
      for (int nb = 0; nb < NB2; ++nb) {
        const int col = nb * 16 + lrow;
        if (col < P.y_cols) {
          if (P.add_inplace) yrow[col] += vals[nb][i];  // row owned by this wave
          else               yrow[col] = vals[nb][i];
        }
      }
    }
  }
}

// ---------------------------------------------------------------------------
// Host orchestration
// ---------------------------------------------------------------------------
extern "C" void kernel_launch(void* const* d_in, const int* in_sizes, int n_in,
                              void* d_out, int out_size, void* d_ws, size_t ws_size,
                              hipStream_t stream) {
  (void)in_sizes; (void)n_in; (void)out_size; (void)ws_size;
  const float* node_feat = (const float*)d_in[0];
  const float* edge_feat = (const float*)d_in[1];
  const float* ctx_feat  = (const float*)d_in[2];
  const int* senders   = (const int*)d_in[3];
  const int* receivers = (const int*)d_in[4];

  // ---- params pytree flattening (dicts sorted by key, lists in order) ----
  // params: dec_ctx, dec_edge, dec_node, enc_ctx, enc_edge, enc_node, proc[0..1]{ctx,edge,node}
  // MLP dict: layers[0].b, layers[0].w, layers[1].b, layers[1].w, (ln_beta, ln_gamma)
  struct MlpIdx { int b0, w0, b1, w1, lnb, lng; };
  int ix = 5;
  auto take = [&](bool ln) {
    MlpIdx m; m.b0 = ix++; m.w0 = ix++; m.b1 = ix++; m.w1 = ix++;
    m.lnb = ln ? ix++ : -1; m.lng = ln ? ix++ : -1; return m;
  };
  MlpIdx dec_ctx = take(false), dec_edge = take(false), dec_node = take(false);
  MlpIdx enc_ctx = take(true), enc_edge = take(true), enc_node = take(true);
  MlpIdx proc_ctx[2], proc_edge[2], proc_node[2];
  for (int s = 0; s < 2; ++s) {
    proc_ctx[s] = take(true); proc_edge[s] = take(true); proc_node[s] = take(true);
  }

  // ---- workspace carve-out (deterministic every call) ----
  char* wsb = (char*)d_ws;
  size_t off = 0;
  auto alloc = [&](size_t bytes) -> char* {
    off = (off + 255) & ~(size_t)255;
    char* p = wsb + off; off += bytes; return p;
  };
  float* e_lat  = (float*)alloc((size_t)GN_EDGES * GN_LATENT * 4);  // 102.4 MB
  float* v_lat  = (float*)alloc((size_t)GN_NODES * GN_LATENT * 4);  // 12.8 MB
  float* c_lat  = (float*)alloc(GN_LATENT * 4);
  float* pooled = (float*)alloc((size_t)GN_NODES * GN_LATENT * 4);  // 12.8 MB
  float* e_sum  = (float*)alloc(GN_LATENT * 4);
  float* v_sum  = (float*)alloc(GN_LATENT * 4);

  struct MlpDev { const float *b1, *b2, *gamma, *beta; const __bf16 *wt1, *wt2; };
  auto setup_w = [&](MlpIdx mi, int kin, int nsrc, int npad) -> MlpDev {
    MlpDev d;
    d.b1 = (const float*)d_in[mi.b0];
    d.b2 = (const float*)d_in[mi.b1];
    d.beta  = mi.lnb >= 0 ? (const float*)d_in[mi.lnb] : nullptr;
    d.gamma = mi.lng >= 0 ? (const float*)d_in[mi.lng] : nullptr;
    __bf16* wt1 = (__bf16*)alloc((size_t)128 * kin * 2);
    __bf16* wt2 = (__bf16*)alloc((size_t)npad * 128 * 2);
    int t1 = 128 * kin;
    convert_wt<<<(t1 + 255) / 256, 256, 0, stream>>>((const float*)d_in[mi.w0], wt1, kin, 128, 128);
    int t2 = npad * 128;
    convert_wt<<<(t2 + 255) / 256, 256, 0, stream>>>((const float*)d_in[mi.w1], wt2, 128, nsrc, npad);
    d.wt1 = wt1; d.wt2 = wt2;
    return d;
  };
  MlpDev w_enc_node = setup_w(enc_node, 32, 64, 64);
  MlpDev w_enc_edge = setup_w(enc_edge, 32, 64, 64);
  MlpDev w_enc_ctx  = setup_w(enc_ctx, 32, 64, 64);
  MlpDev w_pe[2], w_pn[2], w_pc[2];
  for (int s = 0; s < 2; ++s) {
    w_pe[s] = setup_w(proc_edge[s], 256, 64, 64);
    w_pn[s] = setup_w(proc_node[s], 192, 64, 64);
    w_pc[s] = setup_w(proc_ctx[s], 192, 64, 64);
  }
  MlpDev w_dec_edge = setup_w(dec_edge, 64, 8, 16);
  MlpDev w_dec_node = setup_w(dec_node, 64, 8, 16);
  MlpDev w_dec_ctx  = setup_w(dec_ctx, 64, 8, 16);

  auto mkP = [&](MlpDev w, float* y, int y_ld, int y_cols, int addip, int M) -> MlpP {
    MlpP P{};
    P.wt1 = w.wt1; P.b1 = w.b1; P.wt2 = w.wt2; P.b2 = w.b2;
    P.gamma = w.gamma; P.beta = w.beta;
    P.y = y; P.y_ld = y_ld; P.y_cols = y_cols; P.add_inplace = addip;
    P.pool = nullptr; P.pool_idx = nullptr; P.colsum = nullptr; P.M = M;
    return P;
  };
  auto grid_of = [](int M, int nwaves) { return (M + nwaves * 16 - 1) / (nwaves * 16); };

  // ---- encode ----
  {
    MlpP P = mkP(w_enc_node, v_lat, 64, 64, 0, GN_NODES);
    P.piece[0] = {node_feat, nullptr, 32, 0};
    mlp_kernel<32, 64, true><<<grid_of(GN_NODES, 8), 256, 0, stream>>>(P);
  }
  {
    MlpP P = mkP(w_enc_edge, e_lat, 64, 64, 0, GN_EDGES);
    P.piece[0] = {edge_feat, nullptr, 32, 0};
    mlp_kernel<32, 64, true><<<grid_of(GN_EDGES, 8), 256, 0, stream>>>(P);
  }
  {
    MlpP P = mkP(w_enc_ctx, c_lat, 64, 64, 0, 1);
    P.piece[0] = {ctx_feat, nullptr, 32, 0};
    mlp_kernel<32, 64, true><<<1, 32, 0, stream>>>(P);
  }

  // ---- process (2 unshared steps, residual in place) ----
  for (int s = 0; s < 2; ++s) {
    zero_f32<<<(int)(((long)GN_NODES * 64 + 255) / 256), 256, 0, stream>>>(pooled, (long)GN_NODES * 64);
    zero_f32<<<1, 64, 0, stream>>>(e_sum, 64);
    zero_f32<<<1, 64, 0, stream>>>(v_sum, 64);
    {  // edge update: concat(e, v[send], v[recv], c) -> e += e_new; pool & colsum e_new
      MlpP P = mkP(w_pe[s], e_lat, 64, 64, 1, GN_EDGES);
      P.piece[0] = {e_lat, nullptr, 64, 0};
      P.piece[1] = {v_lat, senders, 64, 0};
      P.piece[2] = {v_lat, receivers, 64, 0};
      P.piece[3] = {c_lat, nullptr, 64, 1};
      P.pool = pooled; P.pool_idx = receivers; P.colsum = e_sum;
      mlp_kernel<256, 64, true><<<grid_of(GN_EDGES, 8), 256, 0, stream>>>(P);
    }
    {  // node update: concat(v, pooled, c) -> v += v_new; colsum v_new
      MlpP P = mkP(w_pn[s], v_lat, 64, 64, 1, GN_NODES);
      P.piece[0] = {v_lat, nullptr, 64, 0};
      P.piece[1] = {pooled, nullptr, 64, 0};
      P.piece[2] = {c_lat, nullptr, 64, 1};
      P.colsum = v_sum;
      mlp_kernel<192, 64, true><<<grid_of(GN_NODES, 8), 256, 0, stream>>>(P);
    }
    {  // context update: concat(c, sum(e_new), sum(v_new)) -> c += c_new
      MlpP P = mkP(w_pc[s], c_lat, 64, 64, 1, 1);
      P.piece[0] = {c_lat, nullptr, 64, 0};
      P.piece[1] = {e_sum, nullptr, 64, 0};
      P.piece[2] = {v_sum, nullptr, 64, 0};
      mlp_kernel<192, 64, true><<<1, 32, 0, stream>>>(P);
    }
  }

  // ---- decode (64 -> 128 -> 8, no LN); outputs concatenated flat ----
  float* out = (float*)d_out;
  {
    MlpP P = mkP(w_dec_edge, out, 8, 8, 0, GN_EDGES);
    P.piece[0] = {e_lat, nullptr, 64, 0};
    mlp_kernel<64, 16, false><<<grid_of(GN_EDGES, 8), 256, 0, stream>>>(P);
  }
  {
    MlpP P = mkP(w_dec_node, out + (size_t)GN_EDGES * 8, 8, 8, 0, GN_NODES);
    P.piece[0] = {v_lat, nullptr, 64, 0};
    mlp_kernel<64, 16, false><<<grid_of(GN_NODES, 8), 256, 0, stream>>>(P);
  }
  {
    MlpP P = mkP(w_dec_ctx, out + (size_t)(GN_EDGES + GN_NODES) * 8, 8, 8, 0, 1);
    P.piece[0] = {c_lat, nullptr, 64, 0};
    mlp_kernel<64, 16, false><<<1, 32, 0, stream>>>(P);
  }
}
